// ScaledDotProductAttention_27994596836229
// MI455X (gfx1250) — matrix operand order repaired and checked by reading, compiled, not run on hardware
//
#include <hip/hip_runtime.h>

// Fused scaled-dot-product attention, 32-query tile per block.
// B=8, S=2048, D=512. Outputs: context [B,S,D] then weights [B,S,S], fp32.
// Fast path (ws_size >= 32MB): K -> bf16 rows, V -> bf16 transposed [D][S] in d_ws.

#define BATCH   8
#define SEQ     2048
#define DIM     512
#define MQ      32          // query rows per block
#define THREADS 256
#define VT_STRIDE 40        // padded bf16 stride for slow-path V^T staging
#define TT      64          // transpose tile

typedef __attribute__((ext_vector_type(16))) __bf16 v16bf;
typedef __attribute__((ext_vector_type(8)))  __bf16 v8bf;
typedef __attribute__((ext_vector_type(4)))  __bf16 v4bf;
typedef __attribute__((ext_vector_type(8)))  float  v8f;
typedef __attribute__((ext_vector_type(4)))  float  f32x4;

union AB16 { v16bf v; v8bf h[2]; };

// ---------------- prepass 1: K f32 -> bf16 rows ----------------
__global__ __launch_bounds__(256)
void cvt_k_bf16_kernel(const float* __restrict__ K, __bf16* __restrict__ Kh, int n4) {
    const int i = blockIdx.x * blockDim.x + threadIdx.x;
    if (i >= n4) return;
    f32x4 k = ((const f32x4*)K)[i];
    v4bf kk;
    #pragma unroll
    for (int t = 0; t < 4; ++t) kk[t] = (__bf16)k[t];
    ((v4bf*)Kh)[i] = kk;
}

// ---------------- prepass 2: V f32 [S][D] -> bf16 V^T [D][S] (tiled) ----------------
__global__ __launch_bounds__(256)
void transpose_v_bf16_kernel(const float* __restrict__ V, __bf16* __restrict__ Vt) {
    __shared__ __bf16 lt[TT][TT + 8];
    const int b  = blockIdx.z;
    const int d0 = blockIdx.x * TT;
    const int s0 = blockIdx.y * TT;
    const float* src = V + ((size_t)b * SEQ + s0) * DIM + d0;
    for (int i = threadIdx.x * 4; i < TT * TT; i += 256 * 4) {
        const int r = i >> 6, c = i & 63;
        f32x4 x = *(const f32x4*)(src + (size_t)r * DIM + c);
        v4bf y;
        #pragma unroll
        for (int t = 0; t < 4; ++t) y[t] = (__bf16)x[t];
        *(v4bf*)(&lt[r][c]) = y;
    }
    __syncthreads();
    __bf16* dst = Vt + ((size_t)b * DIM + d0) * SEQ + s0;
    for (int i = threadIdx.x * 4; i < TT * TT; i += 256 * 4) {
        const int d = i >> 6, s = i & 63;
        v4bf y;
        #pragma unroll
        for (int t = 0; t < 4; ++t) y[t] = lt[s + t][d];
        *(v4bf*)(dst + (size_t)d * SEQ + s) = y;
    }
}

// ---------------- fused attention ----------------
template <bool BF16KV>
__global__ __launch_bounds__(THREADS, 1)
void sdpa_kernel(const float* __restrict__ Q,
                 const float* __restrict__ Kf, const float* __restrict__ Vf,
                 const __bf16* __restrict__ Kh, const __bf16* __restrict__ Vth,
                 float* __restrict__ ctx, float* __restrict__ wts) {
    extern __shared__ char smem[];
    __bf16* sc = (__bf16*)smem;                  // 32 x 2048 bf16: scores -> exp -> P (128 KB)
    __bf16* qs = (__bf16*)(smem + MQ * SEQ * 2); // 32 x 512 bf16 Q tile (32 KB, stages 1-2)
    __bf16* vt = (__bf16*)(smem + MQ * SEQ * 2); // slow path: 512 x VT_STRIDE V^T chunk (40 KB)

    const int b     = blockIdx.y;
    const int q0    = blockIdx.x * MQ;
    const int tid   = threadIdx.x;
    const int wave  = tid >> 5;
    const int lane  = tid & 31;
    const int lhalf = lane >> 4;     // 0: lanes 0-15, 1: lanes 16-31
    const int l16   = lane & 15;

    const float scale = 0.044194173824159216f;   // 1/sqrt(512)

    // ---- Stage 1: Q tile -> LDS bf16, pre-scaled ----
    {
        const float* qb = Q + ((size_t)b * SEQ + q0) * DIM;
        for (int i = tid; i < MQ * DIM; i += THREADS)
            qs[i] = (__bf16)(qb[i] * scale);
    }
    __syncthreads();

    // ---- Stage 2: scores = (Q*scale) @ K^T -> LDS bf16 ----
    // n-tiles register-blocked in groups of 4: one A-fragment pair feeds 8 WMMAs.
    {
        const float*  kbf = Kf + (size_t)b * SEQ * DIM;
        const __bf16* kbh = Kh + (size_t)b * SEQ * DIM;
        for (int jg = 0; jg < 4; ++jg) {
            const int ng = wave * 256 + jg * 64;
            v8f acc[4][2] = {{{}, {}}, {{}, {}}, {{}, {}}, {{}, {}}};
            #pragma unroll 2
            for (int k0 = 0; k0 < DIM; k0 += 32) {
                // A fragments for both 16-row M-tiles (shared by all 4 n-tiles)
                AB16 a0, a1;
                const __bf16* ap = qs + l16 * DIM + k0 + (lhalf << 4);
                a0.h[0] = *(const v8bf*)(ap);
                a0.h[1] = *(const v8bf*)(ap + 8);
                a1.h[0] = *(const v8bf*)(ap + 16 * DIM);
                a1.h[1] = *(const v8bf*)(ap + 16 * DIM + 8);
                #pragma unroll
                for (int jj = 0; jj < 4; ++jj) {
                    const int n0 = ng + jj * 16;
                    AB16 bm;
                    if (BF16KV) {
                        const __bf16* krow = kbh + (size_t)(n0 + l16) * DIM + k0 + (lhalf << 4);
                        __builtin_prefetch(krow + 128, 0, 3);
                        bm.h[0] = *(const v8bf*)(krow);
                        bm.h[1] = *(const v8bf*)(krow + 8);
                    } else {
                        const float* krow = kbf + (size_t)(n0 + l16) * DIM + k0 + (lhalf << 4);
                        float fk[16];
                        *(f32x4*)(fk + 0)  = ((const f32x4*)krow)[0];
                        *(f32x4*)(fk + 4)  = ((const f32x4*)krow)[1];
                        *(f32x4*)(fk + 8)  = ((const f32x4*)krow)[2];
                        *(f32x4*)(fk + 12) = ((const f32x4*)krow)[3];
                        #pragma unroll
                        for (int t = 0; t < 16; ++t) bm.v[t] = (__bf16)fk[t];
                    }
                    acc[jj][0] = __builtin_amdgcn_wmma_f32_16x16x32_bf16(
                                     false, a0.v, false, bm.v, (short)0, acc[jj][0], false, false);
                    acc[jj][1] = __builtin_amdgcn_wmma_f32_16x16x32_bf16(
                                     false, a1.v, false, bm.v, (short)0, acc[jj][1], false, false);
                }
            }
            #pragma unroll
            for (int jj = 0; jj < 4; ++jj) {
                const int n0 = ng + jj * 16;
                #pragma unroll
                for (int r = 0; r < 8; ++r) {
                    const int m = r + (lhalf << 3);
                    sc[(size_t)m * SEQ + n0 + l16]        = (__bf16)acc[jj][0][r];
                    sc[(size_t)(m + 16) * SEQ + n0 + l16] = (__bf16)acc[jj][1][r];
                }
            }
        }
    }
    __syncthreads();

    // ---- Stage 3: softmax in place (4 rows per wave, 8-wide vectorized) ----
    {
        for (int rr = 0; rr < 4; ++rr) {
            const int r = wave * 4 + rr;
            __bf16* row = sc + (size_t)r * SEQ;
            float m = -3.402823466e38f;
            for (int c0 = lane * 8; c0 < SEQ; c0 += 256) {
                v8bf x = *(const v8bf*)(row + c0);
                #pragma unroll
                for (int t = 0; t < 8; ++t) m = fmaxf(m, (float)x[t]);
            }
            #pragma unroll
            for (int off = 16; off; off >>= 1) m = fmaxf(m, __shfl_xor(m, off, 32));
            float sum = 0.f;
            for (int c0 = lane * 8; c0 < SEQ; c0 += 256) {
                v8bf x = *(const v8bf*)(row + c0);
                v8bf e;
                #pragma unroll
                for (int t = 0; t < 8; ++t) {
                    float ef = __expf((float)x[t] - m);
                    sum += ef;
                    e[t] = (__bf16)ef;
                }
                *(v8bf*)(row + c0) = e;
            }
            #pragma unroll
            for (int off = 16; off; off >>= 1) sum += __shfl_xor(sum, off, 32);
            const float inv = 1.0f / sum;
            float* wrow = wts + ((size_t)b * SEQ + q0 + r) * SEQ;
            for (int c0 = lane * 8; c0 < SEQ; c0 += 256) {
                v8bf x = *(const v8bf*)(row + c0);
                f32x4 p0, p1;
                v8bf pb;
                #pragma unroll
                for (int t = 0; t < 4; ++t) { p0[t] = (float)x[t] * inv;     pb[t] = (__bf16)p0[t]; }
                #pragma unroll
                for (int t = 0; t < 4; ++t) { p1[t] = (float)x[4 + t] * inv; pb[4 + t] = (__bf16)p1[t]; }
                __builtin_nontemporal_store(p0, (f32x4*)(wrow + c0));       // f32 weights stream
                __builtin_nontemporal_store(p1, (f32x4*)(wrow + c0 + 4));
                *(v8bf*)(row + c0) = pb;                                    // bf16 P for PV
            }
        }
    }
    __syncthreads();

    // ---- Stage 4: context = P @ V ----
    {
        v8f acc[2][4] = {{{}, {}, {}, {}}, {{}, {}, {}, {}}};
        if (BF16KV) {
            // B-fragments straight from pre-transposed bf16 V^T [DIM][SEQ]
            const __bf16* vtb = Vth + (size_t)b * DIM * SEQ;
            #pragma unroll 2
            for (int kt = 0; kt < SEQ / 32; ++kt) {
                const int k0 = kt * 32;
                AB16 a0, a1;
                const __bf16* ap = sc + (size_t)l16 * SEQ + k0 + (lhalf << 3);
                a0.h[0] = *(const v8bf*)(ap);
                a0.h[1] = *(const v8bf*)(ap + 16);
                a1.h[0] = *(const v8bf*)(ap + (size_t)16 * SEQ);
                a1.h[1] = *(const v8bf*)(ap + (size_t)16 * SEQ + 16);
                #pragma unroll
                for (int j = 0; j < 4; ++j) {
                    const int n0 = wave * 64 + j * 16;
                    const __bf16* vrow = vtb + (size_t)(n0 + l16) * SEQ + k0 + (lhalf << 3);
                    __builtin_prefetch(vrow + 128, 0, 3);
                    AB16 bm;
                    bm.h[0] = *(const v8bf*)(vrow);
                    bm.h[1] = *(const v8bf*)(vrow + 16);
                    acc[0][j] = __builtin_amdgcn_wmma_f32_16x16x32_bf16(
                                    false, a0.v, false, bm.v, (short)0, acc[0][j], false, false);
                    acc[1][j] = __builtin_amdgcn_wmma_f32_16x16x32_bf16(
                                    false, a1.v, false, bm.v, (short)0, acc[1][j], false, false);
                }
            }
        } else {
            // fallback: stage f32 V chunks through LDS with transpose
            const float* vbf = Vf + (size_t)b * SEQ * DIM;
            for (int kt = 0; kt < SEQ / 32; ++kt) {
                const int k0 = kt * 32;
                const float* vchunk = vbf + (size_t)k0 * DIM;
                for (int i = tid * 4; i < 32 * DIM; i += THREADS * 4) {
                    f32x4 x = *(const f32x4*)(vchunk + i);
                    const int vrow = i >> 9, vcol = i & (DIM - 1);
                    vt[(vcol + 0) * VT_STRIDE + vrow] = (__bf16)x[0];
                    vt[(vcol + 1) * VT_STRIDE + vrow] = (__bf16)x[1];
                    vt[(vcol + 2) * VT_STRIDE + vrow] = (__bf16)x[2];
                    vt[(vcol + 3) * VT_STRIDE + vrow] = (__bf16)x[3];
                }
                __syncthreads();
                AB16 a0, a1;
                const __bf16* ap = sc + (size_t)l16 * SEQ + k0 + (lhalf << 3);
                a0.h[0] = *(const v8bf*)(ap);
                a0.h[1] = *(const v8bf*)(ap + 16);
                a1.h[0] = *(const v8bf*)(ap + (size_t)16 * SEQ);
                a1.h[1] = *(const v8bf*)(ap + (size_t)16 * SEQ + 16);
                #pragma unroll
                for (int j = 0; j < 4; ++j) {
                    const int n0 = wave * 64 + j * 16;
                    AB16 bm;
                    const __bf16* bp = vt + (n0 + l16) * VT_STRIDE + (lhalf << 3);
                    bm.h[0] = *(const v8bf*)(bp);
                    bm.h[1] = *(const v8bf*)(bp + 16);
                    acc[0][j] = __builtin_amdgcn_wmma_f32_16x16x32_bf16(
                                    false, a0.v, false, bm.v, (short)0, acc[0][j], false, false);
                    acc[1][j] = __builtin_amdgcn_wmma_f32_16x16x32_bf16(
                                    false, a1.v, false, bm.v, (short)0, acc[1][j], false, false);
                }
                __syncthreads();
            }
        }
        // store context tile (write-once stream)
        float* cb = ctx + ((size_t)b * SEQ + q0) * DIM;
        #pragma unroll
        for (int mi = 0; mi < 2; ++mi)
            #pragma unroll
            for (int j = 0; j < 4; ++j) {
                const int n0 = wave * 64 + j * 16;
                #pragma unroll
                for (int r = 0; r < 8; ++r) {
                    const int m = mi * 16 + r + (lhalf << 3);
                    __builtin_nontemporal_store(acc[mi][j][r], cb + (size_t)m * DIM + n0 + l16);
                }
            }
    }
}

extern "C" void kernel_launch(void* const* d_in, const int* in_sizes, int n_in,
                              void* d_out, int out_size, void* d_ws, size_t ws_size,
                              hipStream_t stream) {
    (void)in_sizes; (void)n_in; (void)out_size;
    const float* Q = (const float*)d_in[0];
    const float* K = (const float*)d_in[1];
    const float* V = (const float*)d_in[2];
    float* ctx = (float*)d_out;
    float* wts = ctx + (size_t)BATCH * SEQ * DIM;

    const size_t N    = (size_t)BATCH * SEQ * DIM;                    // 8.4M elems / matrix
    const bool   fast = (ws_size >= 2 * N * sizeof(unsigned short));  // 32 MB scratch

    dim3 grid(SEQ / MQ, BATCH);   // 64 x 8 blocks

    if (fast) {
        __bf16* Kh  = (__bf16*)d_ws;
        __bf16* Vth = Kh + N;
        const int n4 = (int)(N / 4);
        cvt_k_bf16_kernel<<<(n4 + 255) / 256, 256, 0, stream>>>(K, Kh, n4);
        dim3 tgrid(DIM / TT, SEQ / TT, BATCH);
        transpose_v_bf16_kernel<<<tgrid, 256, 0, stream>>>(V, Vth);

        const size_t smem = (size_t)MQ * SEQ * 2 + (size_t)MQ * DIM * 2;  // 160 KB -> 2 blocks/WGP
        hipFuncSetAttribute(reinterpret_cast<const void*>(sdpa_kernel<true>),
                            hipFuncAttributeMaxDynamicSharedMemorySize, (int)smem);
        sdpa_kernel<true><<<grid, THREADS, smem, stream>>>(Q, nullptr, nullptr, Kh, Vth, ctx, wts);
    } else {
        const size_t smem = (size_t)MQ * SEQ * 2 + (size_t)DIM * VT_STRIDE * 2;  // 168 KB
        hipFuncSetAttribute(reinterpret_cast<const void*>(sdpa_kernel<false>),
                            hipFuncAttributeMaxDynamicSharedMemorySize, (int)smem);
        sdpa_kernel<false><<<grid, THREADS, smem, stream>>>(Q, K, V, nullptr, nullptr, ctx, wts);
    }
}
